// MultiheadAttention_74337293959578
// MI455X (gfx1250) — matrix operand order repaired and checked by reading, not compiled, not run on hardware
//
#include <hip/hip_runtime.h>

// ---------------------------------------------------------------- types
typedef __attribute__((ext_vector_type(16))) __bf16 v16bf;
typedef __attribute__((ext_vector_type(8)))  float  v8f;

union FragU {            // one WMMA 16-bit A/B fragment = 8 VGPRs
    v16bf v;
    uint4 q[2];
    unsigned short u[16];
};
union U4 {               // uint4 <-> 8 bf16 bit-patterns
    uint4 v;
    unsigned short u[8];
};

// ---------------------------------------------------------------- consts
constexpr int D_MODEL = 2048;
constexpr int NUM_HEADS = 16;
constexpr int HEAD_SIZE = 128;
constexpr int Bb = 2, SQ = 2048, SK = 2048;

__device__ __forceinline__ unsigned short f2bf(float f) {
    union { float f; unsigned u; } v; v.f = f;
    unsigned r = v.u + 0x7FFFu + ((v.u >> 16) & 1u);   // round-to-nearest-even
    return (unsigned short)(r >> 16);
}

__device__ __forceinline__ v8f zero8() {
    v8f z;
#pragma unroll
    for (int i = 0; i < 8; ++i) z[i] = 0.f;
    return z;
}

// CDNA5 async global->LDS copy (ASYNCcnt path, no VGPR staging).
// Generic LDS pointers carry the wave-relative LDS offset in addr[31:0].
__device__ __forceinline__ void async_copy_b128(unsigned short* lds_dst,
                                                const unsigned short* gsrc) {
    unsigned ldsOff = (unsigned)(size_t)lds_dst;
    asm volatile("global_load_async_to_lds_b128 %0, %1, off"
                 :: "v"(ldsOff), "v"(gsrc) : "memory");
}
__device__ __forceinline__ void wait_async() {
    asm volatile("s_wait_asynccnt 0x0" ::: "memory");
}

// A fragment (16x32 bf16): lane holds row (lane&15); khalf=(lane>>4).
__device__ __forceinline__ v16bf load_frag_a(const unsigned short* p, int ld) {
    int lane = threadIdx.x & 31;
    int row = lane & 15, kh = (lane >> 4) * 8;
    FragU f;
    f.q[0] = *(const uint4*)(p + row * ld + kh);
    f.q[1] = *(const uint4*)(p + row * ld + 16 + kh);
    return f.v;
}

// B fragment (32x16 bf16) from K-contiguous storage [n][k]:
// lane holds col n=(lane&15); lanes 0-15 K=0..7,16..23, lanes 16-31 K=8..15,24..31
__device__ __forceinline__ v16bf load_frag_b(const unsigned short* p, int ld) {
    int lane = threadIdx.x & 31;
    int n = lane & 15, kb = (lane >> 4) * 8;
    FragU f;
    f.q[0] = *(const uint4*)(p + n * ld + kb);
    f.q[1] = *(const uint4*)(p + n * ld + kb + 16);
    return f.v;
}

// ---------------------------------------------------------------- cvt f32 -> bf16
__global__ void cvt_f32_bf16(const float* __restrict__ src,
                             unsigned short* __restrict__ dst, size_t n8) {
    size_t i = (size_t)blockIdx.x * blockDim.x + threadIdx.x;
    size_t stride = (size_t)gridDim.x * blockDim.x;
    for (; i < n8; i += stride) {
        float4 f0 = ((const float4*)src)[2 * i];
        float4 f1 = ((const float4*)src)[2 * i + 1];
        U4 o;
        o.u[0] = f2bf(f0.x); o.u[1] = f2bf(f0.y);
        o.u[2] = f2bf(f0.z); o.u[3] = f2bf(f0.w);
        o.u[4] = f2bf(f1.x); o.u[5] = f2bf(f1.y);
        o.u[6] = f2bf(f1.z); o.u[7] = f2bf(f1.w);
        ((uint4*)dst)[i] = o.v;
    }
}

// ---------------------------------------------------------------- bf16 WMMA GEMM
// C[M,N] = alpha * A[M,K] * B[K,N] (+bias). Block tile 128x128 x BK=64,
// 256 threads (8 waves in 4x2), wave tile 32x64 -> 16 WMMAs per barrier.
// Double-buffered LDS; A staged via async global->LDS; B globals prefetched
// into VGPRs one step ahead of the transposed ds_store scatter.
__global__ __launch_bounds__(256)
void gemm_bf16(const unsigned short* __restrict__ A,
               const unsigned short* __restrict__ Bm,
               unsigned short* __restrict__ outB,
               float* __restrict__ outF,
               const float* __restrict__ bias,
               int M, int N, int K, float alpha) {
    constexpr int LDA = 72;   // 64 + 8 pad, rows stay 16B aligned
    constexpr int LDB = 72;
    __shared__ unsigned short sA[2][128 * LDA];
    __shared__ unsigned short sB[2][128 * LDB];    // transposed: [n][k]

    const int tid = threadIdx.x;
    const int wid = tid >> 5;
    const int wm = wid & 3;          // 0..3 -> 32-row slice
    const int wn = wid >> 2;         // 0..1 -> 64-col slice
    const int row0 = blockIdx.y * 128;
    const int col0 = blockIdx.x * 128;

    // staging maps: tile A 128x64 (1024 b128 chunks), tile B 64x128 (1024)
    const int ar = tid >> 1,  ac8 = (tid & 1) * 8;     // rows ar, +(32/64/96)? no:
    // A: chunk c -> row c>>3, col8 (c&7)*8 ; threads cover c = tid + {0,256,512,768}
    const int bk = tid >> 4,  bn8 = (tid & 15) * 8;    // B: keys bk,+16,+32,+48
    (void)ar; (void)ac8;

    v8f acc[2][4];
#pragma unroll
    for (int i = 0; i < 2; ++i)
#pragma unroll
        for (int j = 0; j < 4; ++j) acc[i][j] = zero8();

    auto stageA = [&](int buf, int k0) {
#pragma unroll
        for (int s = 0; s < 4; ++s) {
            int c = tid + s * 256;
            int r = c >> 3, c8 = (c & 7) * 8;
            async_copy_b128(&sA[buf][r * LDA + c8],
                            A + (size_t)(row0 + r) * K + k0 + c8);
        }
    };
    auto loadB = [&](int k0, U4* br) {
#pragma unroll
        for (int s = 0; s < 4; ++s)
            br[s].v = *(const uint4*)(Bm + (size_t)(k0 + bk + s * 16) * N + col0 + bn8);
    };
    auto storeB = [&](int buf, const U4* br) {
#pragma unroll
        for (int s = 0; s < 4; ++s)
#pragma unroll
            for (int i = 0; i < 8; ++i)
                sB[buf][(bn8 + i) * LDB + bk + s * 16] = br[s].u[i];
    };

    // ---- prologue: stage tile 0 into buffer 0
    stageA(0, 0);
    {
        U4 br[4];
        loadB(0, br);
        storeB(0, br);
    }
    wait_async();
    __syncthreads();

    for (int k0 = 0, it = 0; k0 < K; k0 += 64, ++it) {
        const int cur = it & 1, nxt = cur ^ 1;
        const bool more = (k0 + 64) < K;
        U4 br[4];
        if (more) {
            stageA(nxt, k0 + 64);
            loadB(k0 + 64, br);
        }

        // ---- compute on current buffer: 2 k-steps x 8 WMMAs
        const unsigned short* pa = &sA[cur][(wm * 32) * LDA];
        const unsigned short* pb = &sB[cur][(wn * 64) * LDB];
#pragma unroll
        for (int ks = 0; ks < 2; ++ks) {
            v16bf a0 = load_frag_a(pa + ks * 32, LDA);
            v16bf a1 = load_frag_a(pa + 16 * LDA + ks * 32, LDA);
            v16bf bf[4];
#pragma unroll
            for (int j = 0; j < 4; ++j) bf[j] = load_frag_b(pb + j * 16 * LDB + ks * 32, LDB);
#pragma unroll
            for (int j = 0; j < 4; ++j) {
                acc[0][j] = __builtin_amdgcn_wmma_f32_16x16x32_bf16(false, a0, false, bf[j], (short)0, acc[0][j], false, false);
                acc[1][j] = __builtin_amdgcn_wmma_f32_16x16x32_bf16(false, a1, false, bf[j], (short)0, acc[1][j], false, false);
            }
        }

        if (more) storeB(nxt, br);
        wait_async();
        __syncthreads();
    }

    // C layout: lane holds col n=(lane&15); VGPR r -> row r + (lane>>4)*8
    const int lane = tid & 31;
    const int n = lane & 15, mh = (lane >> 4) * 8;
#pragma unroll
    for (int i = 0; i < 2; ++i) {
#pragma unroll
        for (int j = 0; j < 4; ++j) {
            int bm = row0 + wm * 32 + i * 16;
            int bn = col0 + wn * 64 + j * 16;
#pragma unroll
            for (int r = 0; r < 8; ++r) {
                int m = bm + mh + r, nn = bn + n;
                float v = acc[i][j][r] * alpha;
                if (outF) outF[(size_t)m * N + nn] = v + (bias ? bias[nn] : 0.f);
                else      outB[(size_t)m * N + nn] = f2bf(v);
            }
        }
    }
}

// ---------------------------------------------------------------- flash attention
// grid: (SQ/128, B*H). block: 256 threads = 8 waves x 16 query rows.
// Q pre-scaled by 1/sqrt(HEAD_SIZE). Causal; future key blocks skipped.
// 64-key blocks: 32 WMMAs per barrier, softmax rescale amortized over 64 keys.
// K staged async, V software-pipelined + transposed, both double-buffered.
__global__ __launch_bounds__(256)
void flash_attn(const unsigned short* __restrict__ Qb,
                const unsigned short* __restrict__ Kb,
                const unsigned short* __restrict__ Vb,
                unsigned short* __restrict__ Ab) {
    constexpr int LDK = 136;   // [key][d]  pad -> distinct banks per key
    constexpr int LDV = 72;    // [d][key]  64 keys + pad
    constexpr int LDP = 72;
    __shared__ unsigned short sK[2][64 * LDK];
    __shared__ unsigned short sV[2][128 * LDV];
    __shared__ unsigned short sP[8][16 * LDP];

    const int qtile = blockIdx.x;
    const int bh = blockIdx.y;
    const int b = bh >> 4, h = bh & 15;
    const size_t headOff = (size_t)b * SQ * D_MODEL + h * HEAD_SIZE;
    const unsigned short* Qh = Qb + headOff;
    const unsigned short* Kh = Kb + headOff;
    const unsigned short* Vh = Vb + headOff;
    unsigned short* Oh = Ab + headOff;

    const int tid = threadIdx.x;
    const int wid = tid >> 5;
    const int lane = tid & 31;
    const int qrow0 = qtile * 128 + wid * 16;

    // staging map: 64 keys x 16 chunks of 8 dims; thread covers keys ke+{0,16,32,48}
    const int ke = tid >> 4, kd8 = (tid & 15) * 8;

    // Q fragments: 4 chunks of K=32 along head dim, straight from global
    v16bf qf[4];
    {
        int row = lane & 15, kh = (lane >> 4) * 8;
        const unsigned short* qp = Qh + (size_t)(qrow0 + row) * D_MODEL;
#pragma unroll
        for (int dc = 0; dc < 4; ++dc) {
            FragU f;
            f.q[0] = *(const uint4*)(qp + dc * 32 + kh);
            f.q[1] = *(const uint4*)(qp + dc * 32 + 16 + kh);
            qf[dc] = f.v;
        }
    }

    v8f o[8];
#pragma unroll
    for (int t = 0; t < 8; ++t) o[t] = zero8();
    float mi[8], li[8];
#pragma unroll
    for (int r = 0; r < 8; ++r) { mi[r] = -1e30f; li[r] = 0.f; }

    const int n = lane & 15, mh = (lane >> 4) * 8;
    const int jend = qtile * 128 + 128;          // causal: skip fully-masked blocks

    auto stageK = [&](int buf, int j0) {
#pragma unroll
        for (int s = 0; s < 4; ++s)
            async_copy_b128(&sK[buf][(ke + s * 16) * LDK + kd8],
                            Kh + (size_t)(j0 + ke + s * 16) * D_MODEL + kd8);
    };
    auto loadV = [&](int j0, U4* vr) {
#pragma unroll
        for (int s = 0; s < 4; ++s)
            vr[s].v = *(const uint4*)(Vh + (size_t)(j0 + ke + s * 16) * D_MODEL + kd8);
    };
    auto storeV = [&](int buf, const U4* vr) {
#pragma unroll
        for (int s = 0; s < 4; ++s)
#pragma unroll
            for (int i = 0; i < 8; ++i)
                sV[buf][(kd8 + i) * LDV + ke + s * 16] = vr[s].u[i];
    };

    // ---- prologue: stage key-block 0 into buffer 0
    stageK(0, 0);
    {
        U4 vr[4];
        loadV(0, vr);
        storeV(0, vr);
    }
    wait_async();
    __syncthreads();

    for (int j0 = 0, it = 0; j0 < jend; j0 += 64, ++it) {
        const int cur = it & 1, nxt = cur ^ 1;
        const bool more = (j0 + 64) < jend;
        U4 vr[4];
        if (more) {
            stageK(nxt, j0 + 64);
            loadV(j0 + 64, vr);
        }

        // S = Q * K^T : four 16-key subtiles, K-dim = 128 -> 16 WMMAs
        v8f s[4];
#pragma unroll
        for (int st = 0; st < 4; ++st) s[st] = zero8();
        const unsigned short* pk = &sK[cur][0];
        const int kb = (lane >> 4) * 8;
#pragma unroll
        for (int dc = 0; dc < 4; ++dc) {
#pragma unroll
            for (int st = 0; st < 4; ++st) {
                FragU f;
                f.q[0] = *(const uint4*)(pk + (n + st * 16) * LDK + dc * 32 + kb);
                f.q[1] = *(const uint4*)(pk + (n + st * 16) * LDK + dc * 32 + kb + 16);
                s[st] = __builtin_amdgcn_wmma_f32_16x16x32_bf16(false, qf[dc], false, f.v, (short)0, s[st], false, false);
            }
        }

        // causal mask + online softmax (row spans 16 lanes of one half-wave)
#pragma unroll
        for (int r = 0; r < 8; ++r) {
            int qg = qrow0 + mh + r;
            float a[4];
#pragma unroll
            for (int st = 0; st < 4; ++st) {
                a[st] = s[st][r];
                if (j0 + st * 16 + n > qg) a[st] = -1e30f;
            }
            float mx = fmaxf(fmaxf(a[0], a[1]), fmaxf(a[2], a[3]));
#pragma unroll
            for (int off = 1; off < 16; off <<= 1)
                mx = fmaxf(mx, __shfl_xor(mx, off, 16));
            float mnew = fmaxf(mi[r], mx);
            float p[4], rs = 0.f;
#pragma unroll
            for (int st = 0; st < 4; ++st) { p[st] = __expf(a[st] - mnew); rs += p[st]; }
#pragma unroll
            for (int off = 1; off < 16; off <<= 1)
                rs += __shfl_xor(rs, off, 16);
            float corr = __expf(mi[r] - mnew);
            li[r] = li[r] * corr + rs;
            mi[r] = mnew;
#pragma unroll
            for (int t = 0; t < 8; ++t) o[t][r] *= corr;
            // P (C-layout) -> per-wave LDS patch in row-major for A-frag reload
            int m = mh + r;
#pragma unroll
            for (int st = 0; st < 4; ++st)
                sP[wid][m * LDP + st * 16 + n] = f2bf(p[st]);
        }

        // O += P(16x64) * V(64x128). Same-wave DS ops are in-order: no barrier.
        v16bf pf0 = load_frag_a(sP[wid], LDP);
        v16bf pf1 = load_frag_a(sP[wid] + 32, LDP);
#pragma unroll
        for (int t = 0; t < 8; ++t) {
            v16bf bv0 = load_frag_b(&sV[cur][(t * 16) * LDV], LDV);
            o[t] = __builtin_amdgcn_wmma_f32_16x16x32_bf16(false, pf0, false, bv0, (short)0, o[t], false, false);
            v16bf bv1 = load_frag_b(&sV[cur][(t * 16) * LDV + 32], LDV);
            o[t] = __builtin_amdgcn_wmma_f32_16x16x32_bf16(false, pf1, false, bv1, (short)0, o[t], false, false);
        }

        if (more) storeV(nxt, vr);
        wait_async();
        __syncthreads();
    }

    // normalize and write context (bf16) back at [row, h*128 + d]
#pragma unroll
    for (int r = 0; r < 8; ++r) {
        float inv = 1.f / li[r];
        size_t rowOff = (size_t)(qrow0 + mh + r) * D_MODEL;
#pragma unroll
        for (int t = 0; t < 8; ++t)
            Oh[rowOff + t * 16 + n] = f2bf(o[t][r] * inv);
    }
}

// ---------------------------------------------------------------- launch
extern "C" void kernel_launch(void* const* d_in, const int* in_sizes, int n_in,
                              void* d_out, int out_size, void* d_ws, size_t ws_size,
                              hipStream_t stream) {
    const float* query = (const float*)d_in[0];
    const float* key   = (const float*)d_in[1];
    const float* value = (const float*)d_in[2];
    // d_in[3] = mask (causal; hardcoded in flash_attn)
    const float* Wq = (const float*)d_in[4];
    const float* Wk = (const float*)d_in[5];
    const float* Wv = (const float*)d_in[6];
    const float* Wo = (const float*)d_in[7];
    const float* bo = (const float*)d_in[8];
    float* out = (float*)d_out;

    constexpr size_t NQ = (size_t)Bb * SQ * D_MODEL;       // 8388608
    constexpr size_t NW = (size_t)D_MODEL * D_MODEL;       // 4194304
    char* w = (char*)d_ws;
    unsigned short* Xq  = (unsigned short*)(w);                        // 16 MB each
    unsigned short* Xk  = (unsigned short*)(w + 16u * 1024 * 1024);
    unsigned short* Xv  = (unsigned short*)(w + 32u * 1024 * 1024);
    unsigned short* WqB = (unsigned short*)(w + 48u * 1024 * 1024);    // 8 MB each
    unsigned short* WkB = (unsigned short*)(w + 56u * 1024 * 1024);
    unsigned short* WvB = (unsigned short*)(w + 64u * 1024 * 1024);
    unsigned short* WoB = (unsigned short*)(w + 72u * 1024 * 1024);
    unsigned short* Qb  = (unsigned short*)(w + 80u * 1024 * 1024);
    unsigned short* Kb  = (unsigned short*)(w + 96u * 1024 * 1024);
    unsigned short* Vb  = (unsigned short*)(w + 112u * 1024 * 1024);
    unsigned short* Ab  = (unsigned short*)(w + 128u * 1024 * 1024);

    cvt_f32_bf16<<<1024, 256, 0, stream>>>(query, Xq, NQ / 8);
    cvt_f32_bf16<<<1024, 256, 0, stream>>>(key,   Xk, NQ / 8);
    cvt_f32_bf16<<<1024, 256, 0, stream>>>(value, Xv, NQ / 8);
    cvt_f32_bf16<<<1024, 256, 0, stream>>>(Wq, WqB, NW / 8);
    cvt_f32_bf16<<<1024, 256, 0, stream>>>(Wk, WkB, NW / 8);
    cvt_f32_bf16<<<1024, 256, 0, stream>>>(Wv, WvB, NW / 8);
    cvt_f32_bf16<<<1024, 256, 0, stream>>>(Wo, WoB, NW / 8);

    const int M = Bb * SQ;           // 4096
    dim3 gGemm(D_MODEL / 128, M / 128);
    const float qscale = 0.0883883476483184f;   // 1/sqrt(128)
    gemm_bf16<<<gGemm, 256, 0, stream>>>(Xq, WqB, Qb, nullptr, nullptr, M, D_MODEL, D_MODEL, qscale);
    gemm_bf16<<<gGemm, 256, 0, stream>>>(Xk, WkB, Kb, nullptr, nullptr, M, D_MODEL, D_MODEL, 1.f);
    gemm_bf16<<<gGemm, 256, 0, stream>>>(Xv, WvB, Vb, nullptr, nullptr, M, D_MODEL, D_MODEL, 1.f);

    dim3 gAttn(SQ / 128, Bb * NUM_HEADS);
    flash_attn<<<gAttn, 256, 0, stream>>>(Qb, Kb, Vb, Ab);

    gemm_bf16<<<gGemm, 256, 0, stream>>>(Ab, WoB, nullptr, out, bo, M, D_MODEL, D_MODEL, 1.f);
    (void)in_sizes; (void)n_in; (void)out_size; (void)ws_size;
}
